// GNNLayer_63350767616783
// MI455X (gfx1250) — matrix operand order repaired and checked by reading, not compiled, not run on hardware
//
#include <hip/hip_runtime.h>

typedef __attribute__((ext_vector_type(16))) __bf16 v16bf;
typedef __attribute__((ext_vector_type(8)))  float  v8f;

#define BATCH 512
#define NNODE 32768
#define NOUT  256
#define KSPLIT 8
#define KCHUNK (NNODE / KSPLIT)   // 4096

// ---------------------------------------------------------------- helpers
__device__ __forceinline__ __bf16 f32_to_bf16(float f) {
  unsigned u = __builtin_bit_cast(unsigned, f);
  unsigned r = u + 0x7FFFu + ((u >> 16) & 1u);   // round-to-nearest-even
  unsigned short h = (unsigned short)(r >> 16);
  return __builtin_bit_cast(__bf16, h);
}

template <typename T> __device__ __forceinline__ T cvt_from_f32(float f);
template <> __device__ __forceinline__ float  cvt_from_f32<float>(float f)  { return f; }
template <> __device__ __forceinline__ __bf16 cvt_from_f32<__bf16>(float f) { return f32_to_bf16(f); }

__device__ __forceinline__ void atomic_fadd_agent(float* p, float v) {
  __hip_atomic_fetch_add(p, v, __ATOMIC_RELAXED, __HIP_MEMORY_SCOPE_AGENT);
}

// ---------------------------------------------------------------- zero fill
__global__ void gnn_zero_f4(float4* __restrict__ p, size_t n4) {
  size_t i = (size_t)blockIdx.x * blockDim.x + threadIdx.x;
  size_t stride = (size_t)gridDim.x * blockDim.x;
  float4 z; z.x = z.y = z.z = z.w = 0.f;
  for (; i < n4; i += stride) p[i] = z;
}

// ---------------------------------------------------------------- out = broadcast(bias)
__global__ void gnn_init_out(float* __restrict__ out, const float* __restrict__ bias) {
  int i = (int)blockIdx.x * (int)blockDim.x + (int)threadIdx.x;  // BATCH*NOUT threads
  out[i] = bias[i & (NOUT - 1)];
}

// ---------------------------------------------------------------- tiled transpose (+optional bf16 convert)
// in: [R, C] row-major f32  ->  out: [C, R] row-major OutT.  R, C multiples of 32.
template <typename OutT>
__global__ void gnn_transpose_cvt(const float* __restrict__ in, OutT* __restrict__ out,
                                  int R, int C) {
  __shared__ float tile[32][33];
  int c0 = blockIdx.x * 32;
  int r0 = blockIdx.y * 32;
  int tx = threadIdx.x, ty = threadIdx.y;      // block (32, 8)
#pragma unroll
  for (int i = 0; i < 32; i += 8)
    tile[ty + i][tx] = in[(size_t)(r0 + ty + i) * C + (c0 + tx)];
  __syncthreads();
#pragma unroll
  for (int i = 0; i < 32; i += 8)
    out[(size_t)(c0 + ty + i) * R + (r0 + tx)] = cvt_from_f32<OutT>(tile[tx][ty + i]);
}

// ---------------------------------------------------------------- COO scatter SpMM
// xt:  [NNODE, BATCH] f32 ;  h1t: [NNODE, BATCH] f32 (pre-zeroed)
// one block per nnz entry, threadIdx.x = batch lane (coalesced gather + atomic scatter)
__global__ void __launch_bounds__(BATCH)
gnn_spmm_scatter(const float* __restrict__ xt, const float* __restrict__ vals,
                 const int* __restrict__ rows, const int* __restrict__ cols,
                 float* __restrict__ h1t, int nnz) {
  int e = blockIdx.x;
  if (e >= nnz) return;
  int b = threadIdx.x;
  int r = rows[e];
  int c = cols[e];
  float v = vals[e];
  float contrib = v * xt[(size_t)c * BATCH + b];
  atomic_fadd_agent(&h1t[(size_t)r * BATCH + b], contrib);
}

// ---------------------------------------------------------------- WMMA GEMM (32x32 tile per wave, 8-way K-split)
// A  : h1  [BATCH, NNODE] bf16 row-major
// Bm : Wt  [NNODE, NOUT]  bf16 row-major   (Wt[k][o] = W[o][k])
// out: [BATCH, NOUT] f32, pre-seeded with bias; partials accumulated with f32 atomics.
// 1024 waves: 16 m-groups (32 rows) x 8 o-groups (32 cols) x 8 k-chunks (4096 deep).
// Per k-step: 2 A-frags (4 x 16B loads) + 2 B-frags (32 strided bf16 loads) -> 4 WMMAs (2x2 register tiling).
__global__ void __launch_bounds__(256)
gnn_gemm_bf16_wmma(const __bf16* __restrict__ A, const __bf16* __restrict__ Bm,
                   float* __restrict__ out) {
  const int K = NNODE;
  int wave = (int)blockIdx.x * ((int)blockDim.x >> 5) + ((int)threadIdx.x >> 5);
  int lane = (int)threadIdx.x & 31;
  int kc   = wave & (KSPLIT - 1);        // k-chunk 0..7
  int rest = wave >> 3;                  // 0..127
  int mg   = rest >> 3;                  // m-group 0..15 (32 rows)
  int og   = rest & 7;                   // o-group 0..7  (32 cols)
  int m0 = mg << 5;
  int o0 = og << 5;
  int kbeg = kc * KCHUNK;

  int m = lane & 15;                     // A-fragment row
  int h = lane >> 4;                     // lane half: K sub-block select

  v8f acc00 = {}, acc01 = {}, acc10 = {}, acc11 = {};

  const __bf16* arow0 = A + (size_t)(m0 + m) * K + h * 8 + kbeg;   // rows m0..m0+15
  const __bf16* arow1 = arow0 + (size_t)16 * K;                    // rows m0+16..m0+31
  const __bf16* brow  = Bm + (size_t)(kbeg + h * 8) * NOUT + o0 + m;  // lane holds B column n=lane&15, K from 8h

#pragma unroll 2
  for (int ks = 0; ks < KCHUNK; ks += 32) {
    union { uint4 u[2]; v16bf v; } a0, a1, b0, b1;
    a0.u[0] = *(const uint4*)(arow0 + ks);
    a0.u[1] = *(const uint4*)(arow0 + ks + 16);
    a1.u[0] = *(const uint4*)(arow1 + ks);
    a1.u[1] = *(const uint4*)(arow1 + ks + 16);
    const __bf16* bp = brow + (size_t)ks * NOUT;
#pragma unroll
    for (int s = 0; s < 16; s++) {
      int kk = (s < 8) ? s : (s + 8);
      b0.v[s] = bp[(size_t)kk * NOUT];
      b1.v[s] = bp[(size_t)kk * NOUT + 16];
    }
    acc00 = __builtin_amdgcn_wmma_f32_16x16x32_bf16(false, a0.v, false, b0.v, (short)0, acc00, false, false);
    acc01 = __builtin_amdgcn_wmma_f32_16x16x32_bf16(false, a0.v, false, b1.v, (short)0, acc01, false, false);
    acc10 = __builtin_amdgcn_wmma_f32_16x16x32_bf16(false, a1.v, false, b0.v, (short)0, acc10, false, false);
    acc11 = __builtin_amdgcn_wmma_f32_16x16x32_bf16(false, a1.v, false, b1.v, (short)0, acc11, false, false);
  }

  // C/D layout: lane n = lane&15, VGPR r -> row r + 8*(lane>>4)
  int n = lane & 15;
#pragma unroll
  for (int r = 0; r < 8; r++) {
    int mm = m0 + r + (h << 3);
    atomic_fadd_agent(&out[(size_t)mm * NOUT + (o0 + n)],        acc00[r]);
    atomic_fadd_agent(&out[(size_t)mm * NOUT + (o0 + 16 + n)],   acc01[r]);
    atomic_fadd_agent(&out[(size_t)(mm + 16) * NOUT + (o0 + n)], acc10[r]);
    atomic_fadd_agent(&out[(size_t)(mm + 16) * NOUT + (o0 + 16 + n)], acc11[r]);
  }
}

// ---------------------------------------------------------------- launch
extern "C" void kernel_launch(void* const* d_in, const int* in_sizes, int n_in,
                              void* d_out, int out_size, void* d_ws, size_t ws_size,
                              hipStream_t stream) {
  const float* x    = (const float*)d_in[0];   // [BATCH, 8, 64, 64] = [BATCH, NNODE]
  const float* W    = (const float*)d_in[1];   // [NOUT, NNODE]
  const float* bias = (const float*)d_in[2];   // [NOUT]
  const float* vals = (const float*)d_in[3];
  const int*   rows = (const int*)d_in[4];
  const int*   cols = (const int*)d_in[5];
  int nnz = in_sizes[3];
  float* out = (float*)d_out;

  char* ws = (char*)d_ws;
  const size_t NB = (size_t)NNODE * BATCH;
  float*  xt   = (float*)(ws);                 // [NNODE, BATCH] f32   (64 MB)
  float*  h1t  = (float*)(ws + NB * 4);        // [NNODE, BATCH] f32   (64 MB)
  __bf16* h1bf = (__bf16*)(ws);                // [BATCH, NNODE] bf16  (reuses xt, 32 MB)
  __bf16* Wt   = (__bf16*)(ws + NB * 8);       // [NNODE, NOUT]  bf16  (16 MB)

  // 1. zero accumulator h1t
  gnn_zero_f4<<<4096, 256, 0, stream>>>((float4*)h1t, NB / 4);

  // 2. transpose x [BATCH, NNODE] -> xt [NNODE, BATCH]
  gnn_transpose_cvt<float><<<dim3(NNODE / 32, BATCH / 32), dim3(32, 8), 0, stream>>>(
      x, xt, BATCH, NNODE);

  // 3. COO scatter SpMM (coalesced over batch)
  gnn_spmm_scatter<<<nnz, BATCH, 0, stream>>>(xt, vals, rows, cols, h1t, nnz);

  // 4. h1t [NNODE, BATCH] f32 -> h1bf [BATCH, NNODE] bf16
  gnn_transpose_cvt<__bf16><<<dim3(BATCH / 32, NNODE / 32), dim3(32, 8), 0, stream>>>(
      h1t, h1bf, NNODE, BATCH);

  // 5. W [NOUT, NNODE] f32 -> Wt [NNODE, NOUT] bf16
  gnn_transpose_cvt<__bf16><<<dim3(NNODE / 32, NOUT / 32), dim3(32, 8), 0, stream>>>(
      W, Wt, NOUT, NNODE);

  // 6. seed out with bias, then WMMA GEMM partial sums (1024 waves = 128 blocks x 8 waves)
  gnn_init_out<<<(BATCH * NOUT) / 256, 256, 0, stream>>>(out, bias);
  gnn_gemm_bf16_wmma<<<128, 256, 0, stream>>>(h1bf, Wt, out);
}
